// GRUPolicy_66769561583648
// MI455X (gfx1250) — compile-verified, matrix operand order checked by reading, not run on hardware
//
#include <hip/hip_runtime.h>

typedef __attribute__((ext_vector_type(8)))  float   v8f;
typedef __attribute__((ext_vector_type(16))) __bf16  v16bf;
typedef __attribute__((ext_vector_type(8)))  __bf16  v8bf;

#define B_TOT      2048
#define STATE_DIM  32
#define REF_DIM    16
#define T_STEPS    100
#define HID        256
#define ACT_DIM    2
#define OBS_STRIDE (STATE_DIM + T_STEPS*REF_DIM)   // 1632
#define ROWS       32                               // batch rows per workgroup
#define NTHREADS   512                              // 16 waves (wave32)

// bf16 weight scratch layout in d_ws (element offsets)
#define WIH0_ELEMS (768*64)        // K padded 48 -> 64
#define WHH_ELEMS  (768*256)
#define WIH0_OFF   0
#define WHH0_OFF   (WIH0_ELEMS)
#define WIH1_OFF   (WHH0_OFF + WHH_ELEMS)
#define WHH1_OFF   (WIH1_OFF + WHH_ELEMS)
#define TOTAL_W    (WHH1_OFF + WHH_ELEMS)

// ---------------- weight f32 -> bf16 conversion (runs once, tiny) -------------
__global__ void cvt_weights(const float* __restrict__ wih0,
                            const float* __restrict__ whh0,
                            const float* __restrict__ wih1,
                            const float* __restrict__ whh1,
                            __bf16* __restrict__ dst) {
  int idx = blockIdx.x * blockDim.x + threadIdx.x;
  if (idx >= TOTAL_W) return;
  float v;
  if (idx < WIH0_ELEMS) {                 // 768 x 64, zero-padded from 768 x 48
    int g = idx >> 6, k = idx & 63;
    v = (k < 48) ? wih0[g*48 + k] : 0.0f;
  } else if (idx < WIH1_OFF) {
    v = whh0[idx - WHH0_OFF];
  } else if (idx < WHH1_OFF) {
    v = wih1[idx - WIH1_OFF];
  } else {
    v = whh1[idx - WHH1_OFF];
  }
  dst[idx] = (__bf16)v;
}

// ---------------- fragment loaders (ISA 16-bit A/B layout) --------------------
// lane<16: elements 0..7 = K[k0..k0+7],  8..15 = K[k0+16..k0+23]
// lane>=16: elements 0..7 = K[k0+8..k0+15], 8..15 = K[k0+24..k0+31]
__device__ __forceinline__ v16bf load_frag(const __bf16* __restrict__ base,
                                           int row, int stride, int k0, int half) {
  const __bf16* p = base + row*stride + k0 + half*8;
  v8bf lo = *(const v8bf*)(p);
  v8bf hi = *(const v8bf*)(p + 16);
  v16bf r;
#pragma unroll
  for (int i = 0; i < 8; ++i) { r[i] = lo[i]; r[i+8] = hi[i]; }
  return r;
}

__device__ __forceinline__ v8f bf_wmma(v16bf a, v16bf b, v8f c) {
  return __builtin_amdgcn_wmma_f32_16x16x32_bf16(false, a, false, b,
                                                 (short)0, c, false, false);
}

__device__ __forceinline__ float sigmoidf_(float x) { return 1.0f / (1.0f + __expf(-x)); }

__device__ __forceinline__ v8f splat8(float x) {
  v8f v;
#pragma unroll
  for (int i = 0; i < 8; ++i) v[i] = x;
  return v;
}

// ---------------- fused persistent GRU policy kernel --------------------------
__global__ __launch_bounds__(NTHREADS, 1)
void gru_policy(const float* __restrict__ obs,
                const float* __restrict__ b_ih0, const float* __restrict__ b_hh0,
                const float* __restrict__ b_ih1, const float* __restrict__ b_hh1,
                const float* __restrict__ fc_w,  const float* __restrict__ fc_b,
                const float* __restrict__ act_hi, const float* __restrict__ act_lo,
                const __bf16* __restrict__ wbf,
                float* __restrict__ out)
{
  __shared__ __align__(16) __bf16 sh_x[ROWS*64];            // x_t: 32 rows x 64 (K-pad)
  __shared__ __align__(16) __bf16 sh_h0[2][ROWS*HID];       // double-buffered h0
  __shared__ __align__(16) __bf16 sh_h1[2][ROWS*HID];       // double-buffered h1
  __shared__ float s_brz0[2*HID], s_bxn0[HID], s_bhn0[HID];
  __shared__ float s_brz1[2*HID], s_bxn1[HID], s_bhn1[HID];

  const __bf16* wih0 = wbf + WIH0_OFF;
  const __bf16* whh0 = wbf + WHH0_OFF;
  const __bf16* wih1 = wbf + WIH1_OFF;
  const __bf16* whh1 = wbf + WHH1_OFF;

  const int tid  = threadIdx.x;
  const int wv   = tid >> 5;        // wave id 0..15 -> gate column tile
  const int lane = tid & 31;
  const int half = lane >> 4;
  const int ln   = lane & 15;
  const int colg = (wv << 4) + ln;  // 0..255 gate-relative column this lane owns
  const int b0   = blockIdx.x * ROWS;

  // --- init biases into LDS ---
  if (tid < 2*HID) {
    s_brz0[tid] = b_ih0[tid] + b_hh0[tid];
    s_brz1[tid] = b_ih1[tid] + b_hh1[tid];
  }
  if (tid < HID) {
    s_bxn0[tid] = b_ih0[2*HID + tid]; s_bhn0[tid] = b_hh0[2*HID + tid];
    s_bxn1[tid] = b_ih1[2*HID + tid]; s_bhn1[tid] = b_hh1[2*HID + tid];
  }
  // --- zero hidden states (both buffers) ---
  {
    __bf16* p0z = &sh_h0[0][0];
    __bf16* p1z = &sh_h1[0][0];
    for (int i = tid; i < 2*ROWS*HID; i += NTHREADS) {
      p0z[i] = (__bf16)0.0f;
      p1z[i] = (__bf16)0.0f;
    }
  }
  // --- x_t: state part (cols 0..31, constant over t) + zero pad (48..63) ---
  for (int i = tid; i < ROWS*STATE_DIM; i += NTHREADS) {
    int r = i >> 5, c = i & 31;
    sh_x[r*64 + c] = (__bf16)obs[(size_t)(b0 + r)*OBS_STRIDE + c];
  }
  for (int i = tid; i < ROWS*16; i += NTHREADS) {
    int r = i >> 4, c = i & 15;
    sh_x[r*64 + 48 + c] = (__bf16)0.0f;
  }
  __syncthreads();

  // per-lane bias scalars
  const float bv_r0  = s_brz0[colg],       bv_z0 = s_brz0[HID + colg];
  const float bv_xn0 = s_bxn0[colg],       bv_hn0 = s_bhn0[colg];
  const float bv_r1  = s_brz1[colg],       bv_z1 = s_brz1[HID + colg];
  const float bv_xn1 = s_bxn1[colg],       bv_hn1 = s_bhn1[colg];

  int p0 = 0, p1 = 0;

  for (int t = 0; t < T_STEPS; ++t) {
    // --- refresh traj part of x_t (cols 32..47); 512 values == 512 threads ---
    {
      int r = tid >> 4, c = tid & 15;
      sh_x[r*64 + 32 + c] =
        (__bf16)obs[(size_t)(b0 + r)*OBS_STRIDE + STATE_DIM + t*REF_DIM + c];
    }
    __syncthreads();   // x_t visible; h1 writes from prev step visible

    // ======================= LAYER 0 =======================
    v8f ar[2], az[2], axn[2], ahn[2];
#pragma unroll
    for (int m = 0; m < 2; ++m) {
      ar[m] = splat8(bv_r0); az[m] = splat8(bv_z0);
      axn[m] = splat8(bv_xn0); ahn[m] = splat8(bv_hn0);
    }
    // x_t @ W_ih0^T  (K = 64 padded, 2 k-tiles)  -> r, z, xn
#pragma unroll 2
    for (int kt = 0; kt < 2; ++kt) {
      int k0 = kt*32;
      v16bf a0 = load_frag(sh_x, ln,      64, k0, half);
      v16bf a1 = load_frag(sh_x, 16 + ln, 64, k0, half);
      v16bf bR = load_frag(wih0, colg,         64, k0, half);
      v16bf bZ = load_frag(wih0, HID + colg,   64, k0, half);
      v16bf bN = load_frag(wih0, 2*HID + colg, 64, k0, half);
      ar[0]  = bf_wmma(a0, bR, ar[0]);   ar[1]  = bf_wmma(a1, bR, ar[1]);
      az[0]  = bf_wmma(a0, bZ, az[0]);   az[1]  = bf_wmma(a1, bZ, az[1]);
      axn[0] = bf_wmma(a0, bN, axn[0]);  axn[1] = bf_wmma(a1, bN, axn[1]);
    }
    // h0 @ W_hh0^T  (K = 256, 8 k-tiles)  -> r, z, hn
#pragma unroll 2
    for (int kt = 0; kt < 8; ++kt) {
      int k0 = kt*32;
      v16bf a0 = load_frag(sh_h0[p0], ln,      HID, k0, half);
      v16bf a1 = load_frag(sh_h0[p0], 16 + ln, HID, k0, half);
      v16bf bR = load_frag(whh0, colg,         HID, k0, half);
      v16bf bZ = load_frag(whh0, HID + colg,   HID, k0, half);
      v16bf bN = load_frag(whh0, 2*HID + colg, HID, k0, half);
      ar[0]  = bf_wmma(a0, bR, ar[0]);   ar[1]  = bf_wmma(a1, bR, ar[1]);
      az[0]  = bf_wmma(a0, bZ, az[0]);   az[1]  = bf_wmma(a1, bZ, az[1]);
      ahn[0] = bf_wmma(a0, bN, ahn[0]);  ahn[1] = bf_wmma(a1, bN, ahn[1]);
    }
    // gates + h0 update (r/z/n columns aligned within this wave)
#pragma unroll
    for (int m = 0; m < 2; ++m) {
#pragma unroll
      for (int i = 0; i < 8; ++i) {
        int row = (m << 4) + (half << 3) + i;
        float r = sigmoidf_(ar[m][i]);
        float z = sigmoidf_(az[m][i]);
        float n = tanhf(axn[m][i] + r*ahn[m][i]);
        float hold = (float)sh_h0[p0][row*HID + colg];
        sh_h0[p0 ^ 1][row*HID + colg] = (__bf16)((1.0f - z)*n + z*hold);
      }
    }
    p0 ^= 1;
    __syncthreads();   // new h0 visible for layer 1

    // ======================= LAYER 1 =======================
#pragma unroll
    for (int m = 0; m < 2; ++m) {
      ar[m] = splat8(bv_r1); az[m] = splat8(bv_z1);
      axn[m] = splat8(bv_xn1); ahn[m] = splat8(bv_hn1);
    }
    // pass 1: h0_new @ W_ih1^T  -> r, z, xn
#pragma unroll 2
    for (int kt = 0; kt < 8; ++kt) {
      int k0 = kt*32;
      v16bf a0 = load_frag(sh_h0[p0], ln,      HID, k0, half);
      v16bf a1 = load_frag(sh_h0[p0], 16 + ln, HID, k0, half);
      v16bf bR = load_frag(wih1, colg,         HID, k0, half);
      v16bf bZ = load_frag(wih1, HID + colg,   HID, k0, half);
      v16bf bN = load_frag(wih1, 2*HID + colg, HID, k0, half);
      ar[0]  = bf_wmma(a0, bR, ar[0]);   ar[1]  = bf_wmma(a1, bR, ar[1]);
      az[0]  = bf_wmma(a0, bZ, az[0]);   az[1]  = bf_wmma(a1, bZ, az[1]);
      axn[0] = bf_wmma(a0, bN, axn[0]);  axn[1] = bf_wmma(a1, bN, axn[1]);
    }
    // pass 2: h1_old @ W_hh1^T  -> r, z, hn
#pragma unroll 2
    for (int kt = 0; kt < 8; ++kt) {
      int k0 = kt*32;
      v16bf c0 = load_frag(sh_h1[p1], ln,      HID, k0, half);
      v16bf c1 = load_frag(sh_h1[p1], 16 + ln, HID, k0, half);
      v16bf dR = load_frag(whh1, colg,         HID, k0, half);
      v16bf dZ = load_frag(whh1, HID + colg,   HID, k0, half);
      v16bf dN = load_frag(whh1, 2*HID + colg, HID, k0, half);
      ar[0]  = bf_wmma(c0, dR, ar[0]);   ar[1]  = bf_wmma(c1, dR, ar[1]);
      az[0]  = bf_wmma(c0, dZ, az[0]);   az[1]  = bf_wmma(c1, dZ, az[1]);
      ahn[0] = bf_wmma(c0, dN, ahn[0]);  ahn[1] = bf_wmma(c1, dN, ahn[1]);
    }
#pragma unroll
    for (int m = 0; m < 2; ++m) {
#pragma unroll
      for (int i = 0; i < 8; ++i) {
        int row = (m << 4) + (half << 3) + i;
        float r = sigmoidf_(ar[m][i]);
        float z = sigmoidf_(az[m][i]);
        float n = tanhf(axn[m][i] + r*ahn[m][i]);
        float hold = (float)sh_h1[p1][row*HID + colg];
        sh_h1[p1 ^ 1][row*HID + colg] = (__bf16)((1.0f - z)*n + z*hold);
      }
    }
    p1 ^= 1;
    // no barrier needed here: next step's two barriers order all h1 accesses
  }
  __syncthreads();

  // ---------------- FC head + tanh squash (ACT_DIM = 2, tiny) ----------------
  if (tid < ROWS*ACT_DIM) {
    int r = tid >> 1, a = tid & 1;
    float acc = fc_b[a];
    for (int k = 0; k < HID; ++k)
      acc += (float)sh_h1[p1][r*HID + k] * fc_w[a*HID + k];
    float hi = act_hi[a], lo = act_lo[a];
    out[(size_t)(b0 + r)*ACT_DIM + a] =
        0.5f*(hi - lo)*tanhf(acc) + 0.5f*(hi + lo);
  }
}

// ---------------- host launch --------------------------------------------------
extern "C" void kernel_launch(void* const* d_in, const int* in_sizes, int n_in,
                              void* d_out, int out_size, void* d_ws, size_t ws_size,
                              hipStream_t stream) {
  const float* obs  = (const float*)d_in[0];
  const float* wih0 = (const float*)d_in[1];
  const float* whh0 = (const float*)d_in[2];
  const float* bih0 = (const float*)d_in[3];
  const float* bhh0 = (const float*)d_in[4];
  const float* wih1 = (const float*)d_in[5];
  const float* whh1 = (const float*)d_in[6];
  const float* bih1 = (const float*)d_in[7];
  const float* bhh1 = (const float*)d_in[8];
  const float* fcw  = (const float*)d_in[9];
  const float* fcb  = (const float*)d_in[10];
  const float* ahi  = (const float*)d_in[11];
  const float* alo  = (const float*)d_in[12];

  __bf16* wbf = (__bf16*)d_ws;   // ~1.22 MB of bf16 weights

  cvt_weights<<<(TOTAL_W + 255)/256, 256, 0, stream>>>(wih0, whh0, wih1, whh1, wbf);
  gru_policy<<<B_TOT/ROWS, NTHREADS, 0, stream>>>(obs, bih0, bhh0, bih1, bhh1,
                                                  fcw, fcb, ahi, alo, wbf,
                                                  (float*)d_out);
}
